// MyGIN_18545668784366
// MI455X (gfx1250) — compile-verified, matrix operand order checked by reading, not run on hardware
//
#include <hip/hip_runtime.h>
#include <hip/hip_bf16.h>

// ---------------------------------------------------------------------------
// GIN forward for MI455X (gfx1250): bf16 WMMA GEMMs + atomic scatter-add.
// ---------------------------------------------------------------------------

#define NUM_NODES 100000
#define NUM_EDGES 1600000
#define NUM_GRAPHS 512
#define HIDDEN 128
#define LAYERS 3
#define NUM_CLASSES 10
#define BN_EPS 1e-5f

typedef __bf16 bf16_t;
typedef bf16_t v16bf __attribute__((ext_vector_type(16)));
typedef bf16_t v8bf  __attribute__((ext_vector_type(8)));
typedef float  v8f   __attribute__((ext_vector_type(8)));

// ---------------------------------------------------------------------------
// Utility: zero a float buffer
// ---------------------------------------------------------------------------
__global__ void zero_f32_kernel(float* __restrict__ p, int n) {
    int i = blockIdx.x * blockDim.x + threadIdx.x;
    if (i < n) p[i] = 0.0f;
}

// ---------------------------------------------------------------------------
// Pack weights [LAYERS][128][128] (k-major, W[k][n]) into WMMA B-operand
// layout: packed[mat][kt][nt][lane][j], 16 bf16 per lane (32B contiguous).
// Lane l holds column n = nt*16 + (l&15);
// half j maps to k = kt*32 + (l>=16 ? 8:0) + (j>=8 ? 16:0) + (j&7).
// ---------------------------------------------------------------------------
__global__ void pack_weights_kernel(const float* __restrict__ W1,
                                    const float* __restrict__ W2,
                                    bf16_t* __restrict__ packed) {
    int idx = blockIdx.x * blockDim.x + threadIdx.x;   // 6*4*8*32*16 = 98304
    if (idx >= 6 * 16384) return;
    int j    = idx & 15;
    int lane = (idx >> 4) & 31;
    int nt   = (idx >> 9) & 7;
    int kt   = (idx >> 12) & 3;
    int m    = idx >> 14;                // 0..5  (layer*2 + which)
    int n = nt * 16 + (lane & 15);
    int k = kt * 32 + ((lane >= 16) ? 8 : 0) + ((j >= 8) ? 16 : 0) + (j & 7);
    int layer = m >> 1;
    const float* W = (m & 1) ? W2 : W1;
    float v = W[((size_t)layer * HIDDEN + k) * HIDDEN + n];
    packed[idx] = (bf16_t)v;
}

// ---------------------------------------------------------------------------
// Scatter-add: agg[dst] += h[src].  One wave per edge, 4 channels per lane.
// ---------------------------------------------------------------------------
__global__ __launch_bounds__(256)
void scatter_add_kernel(const float* __restrict__ h,
                        const int* __restrict__ ei,
                        float* __restrict__ agg, int nedges) {
    int gid  = blockIdx.x * blockDim.x + threadIdx.x;
    int e    = gid >> 5;
    int lane = gid & 31;
    if (e >= nedges) return;
    int src = ei[e];
    int dst = ei[nedges + e];
    const float4 v = *(const float4*)(h + (size_t)src * HIDDEN + lane * 4);
    float* p = agg + (size_t)dst * HIDDEN + lane * 4;
    atomicAdd(p + 0, v.x);
    atomicAdd(p + 1, v.y);
    atomicAdd(p + 2, v.z);
    atomicAdd(p + 3, v.w);
}

// ---------------------------------------------------------------------------
// zin = (1+eps)*h + agg, converted to bf16 for the WMMA A operand.
// ---------------------------------------------------------------------------
__global__ void zin_kernel(const float* __restrict__ h,
                           const float* __restrict__ agg,
                           const float* __restrict__ eps_gin, int layer,
                           bf16_t* __restrict__ zb, int total) {
    int i = blockIdx.x * blockDim.x + threadIdx.x;
    if (i >= total) return;
    float e1 = 1.0f + eps_gin[layer];
    zb[i] = (bf16_t)(e1 * h[i] + agg[i]);
}

// ---------------------------------------------------------------------------
// WMMA GEMM: out = relu(in[nrows,128] @ W[128,128] + bias)
// 8 waves/block, 16 rows per wave, 128 cols per wave.
// Weights staged in LDS in pre-packed B-operand layout.
// ---------------------------------------------------------------------------
template <bool OUT_BF16>
__global__ __launch_bounds__(256)
void gemm128_wmma_kernel(const bf16_t* __restrict__ in,
                         const bf16_t* __restrict__ wpack,
                         const float* __restrict__ bias,
                         void* __restrict__ out, int nrows) {
    __shared__ bf16_t Wl[16384];   // 32 KB
    {
        const int4* s = (const int4*)wpack;
        int4* d = (int4*)Wl;
        #pragma unroll
        for (int i = 0; i < 8; ++i)
            d[threadIdx.x + i * 256] = s[threadIdx.x + i * 256];
    }
    __syncthreads();

    const int wave = threadIdx.x >> 5;
    const int lane = threadIdx.x & 31;
    const int rowBase = (blockIdx.x * 8 + wave) * 16;
    if (rowBase >= nrows) return;          // wave-uniform: EXEC stays all-1s

    const int m     = lane & 15;           // row (A) / col (B,C,D) within tile
    const int khalf = (lane >> 4) << 3;    // 0 or 8

    v8f acc[8];
    #pragma unroll
    for (int nt = 0; nt < 8; ++nt) {
        float bv = bias[nt * 16 + m];      // column is fixed per lane
        acc[nt] = (v8f){bv, bv, bv, bv, bv, bv, bv, bv};
    }

    const bf16_t* arow = in + (size_t)(rowBase + m) * HIDDEN;

    #pragma unroll
    for (int kt = 0; kt < 4; ++kt) {
        const int kb = kt * 32 + khalf;
        v8bf lo = *(const v8bf*)(arow + kb);
        v8bf hi = *(const v8bf*)(arow + kb + 16);
        v16bf a;
        #pragma unroll
        for (int i = 0; i < 8; ++i) { a[i] = lo[i]; a[i + 8] = hi[i]; }
        #pragma unroll
        for (int nt = 0; nt < 8; ++nt) {
            v16bf b = *(const v16bf*)(&Wl[(((kt * 8 + nt) * 32) + lane) * 16]);
            acc[nt] = __builtin_amdgcn_wmma_f32_16x16x32_bf16(
                false, a, false, b, (short)0, acc[nt], false, false);
        }
    }

    // Epilogue: relu + store. Lane covers col c; VGPR i -> row rowOff+i.
    const int rowOff = (lane >> 4) << 3;   // 0 or 8
    #pragma unroll
    for (int nt = 0; nt < 8; ++nt) {
        const int c = nt * 16 + m;
        #pragma unroll
        for (int i = 0; i < 8; ++i) {
            float v = fmaxf(acc[nt][i], 0.0f);
            size_t off = (size_t)(rowBase + rowOff + i) * HIDDEN + c;
            if (OUT_BF16) ((bf16_t*)out)[off] = (bf16_t)v;
            else          ((float*)out)[off]  = v;
        }
    }
}

// ---------------------------------------------------------------------------
// BatchNorm statistics: per-channel sum and sum-of-squares (atomics).
// ---------------------------------------------------------------------------
__global__ void bn_stats_kernel(const float* __restrict__ z,
                                float* __restrict__ stats, int n) {
    int c = threadIdx.x;                   // 128 threads
    float s = 0.0f, ss = 0.0f;
    for (int r = blockIdx.x; r < n; r += gridDim.x) {
        float v = z[(size_t)r * HIDDEN + c];
        s += v;
        ss += v * v;
    }
    atomicAdd(&stats[c], s);
    atomicAdd(&stats[HIDDEN + c], ss);
}

// ---------------------------------------------------------------------------
// BatchNorm apply -> h (next layer input) and hcat[:, l*128:(l+1)*128]
// ---------------------------------------------------------------------------
__global__ void bn_apply_kernel(const float* __restrict__ z,
                                const float* __restrict__ stats,
                                const float* __restrict__ gamma,
                                const float* __restrict__ beta, int layer,
                                float* __restrict__ h,
                                float* __restrict__ hcat, int n) {
    int i = blockIdx.x * blockDim.x + threadIdx.x;
    int total = n * HIDDEN;
    if (i >= total) return;
    int c = i & (HIDDEN - 1);
    int row = i >> 7;
    float invn = 1.0f / (float)n;
    float mu = stats[c] * invn;
    float var = stats[HIDDEN + c] * invn - mu * mu;
    float val = (z[i] - mu) * rsqrtf(var + BN_EPS)
              * gamma[layer * HIDDEN + c] + beta[layer * HIDDEN + c];
    h[i] = val;
    hcat[(size_t)row * (LAYERS * HIDDEN) + layer * HIDDEN + c] = val;
}

// ---------------------------------------------------------------------------
// Global mean pool: atomic accumulate per-graph sums + counts.
// ---------------------------------------------------------------------------
__global__ void pool_acc_kernel(const float* __restrict__ hcat,
                                const int* __restrict__ batch,
                                float* __restrict__ pools,
                                float* __restrict__ counts, int n) {
    long long i = (long long)blockIdx.x * blockDim.x + threadIdx.x;
    long long total = (long long)n * (LAYERS * HIDDEN);
    if (i >= total) return;
    int row = (int)(i / (LAYERS * HIDDEN));
    int c   = (int)(i % (LAYERS * HIDDEN));
    int g = batch[row];
    atomicAdd(&pools[(size_t)g * (LAYERS * HIDDEN) + c], hcat[i]);
    if (c == 0) atomicAdd(&counts[g], 1.0f);
}

// ---------------------------------------------------------------------------
// Head MLP part 1: gm1 = relu(mean_pool @ lin1_w + lin1_b)  [512 x 128]
// ---------------------------------------------------------------------------
__global__ void mlp1_kernel(const float* __restrict__ pools,
                            const float* __restrict__ counts,
                            const float* __restrict__ w,   // [384,128]
                            const float* __restrict__ b,
                            float* __restrict__ gm1) {
    int g = blockIdx.x;        // 512
    int o = threadIdx.x;       // 128
    float inv = 1.0f / fmaxf(counts[g], 1.0f);
    float acc = b[o];
    const float* prow = pools + (size_t)g * (LAYERS * HIDDEN);
    #pragma unroll 4
    for (int k = 0; k < LAYERS * HIDDEN; ++k)
        acc += prow[k] * inv * w[(size_t)k * HIDDEN + o];
    gm1[(size_t)g * HIDDEN + o] = fmaxf(acc, 0.0f);
}

// ---------------------------------------------------------------------------
// Head MLP part 2: out = gm1 @ lin2_w + lin2_b  [512 x 10]
// ---------------------------------------------------------------------------
__global__ void mlp2_kernel(const float* __restrict__ gm1,
                            const float* __restrict__ w,   // [128,10]
                            const float* __restrict__ b,
                            float* __restrict__ out) {
    int g = blockIdx.x;        // 512
    int o = threadIdx.x;       // 32 threads, 10 active
    if (o >= NUM_CLASSES) return;
    float acc = b[o];
    const float* row = gm1 + (size_t)g * HIDDEN;
    #pragma unroll 4
    for (int k = 0; k < HIDDEN; ++k)
        acc += row[k] * w[(size_t)k * NUM_CLASSES + o];
    out[(size_t)g * NUM_CLASSES + o] = acc;
}

// ---------------------------------------------------------------------------
// Host-side orchestration
// ---------------------------------------------------------------------------
static inline size_t align256(size_t x) { return (x + 255) & ~(size_t)255; }

extern "C" void kernel_launch(void* const* d_in, const int* in_sizes, int n_in,
                              void* d_out, int out_size, void* d_ws, size_t ws_size,
                              hipStream_t stream) {
    (void)in_sizes; (void)n_in; (void)out_size; (void)ws_size;

    const float* x        = (const float*)d_in[0];
    const int*   ei       = (const int*)d_in[1];
    const int*   batch    = (const int*)d_in[2];
    const float* W1       = (const float*)d_in[3];
    const float* b1       = (const float*)d_in[4];
    const float* W2       = (const float*)d_in[5];
    const float* b2       = (const float*)d_in[6];
    const float* gamma    = (const float*)d_in[7];
    const float* beta     = (const float*)d_in[8];
    const float* eps_gin  = (const float*)d_in[9];
    const float* lin1_w   = (const float*)d_in[10];
    const float* lin1_b   = (const float*)d_in[11];
    const float* lin2_w   = (const float*)d_in[12];
    const float* lin2_b   = (const float*)d_in[13];
    float* out = (float*)d_out;

    const int N = NUM_NODES, E = NUM_EDGES;
    const size_t NH  = (size_t)N * HIDDEN;            // 12.8M elems
    const size_t NHC = (size_t)N * LAYERS * HIDDEN;   // 38.4M elems

    // Workspace layout
    char* ws = (char*)d_ws;
    size_t off = 0;
    float*  agg    = (float*)(ws + off);  off = align256(off + NH * 4);
    float*  h      = (float*)(ws + off);  off = align256(off + NH * 4);
    float*  z2     = (float*)(ws + off);  off = align256(off + NH * 4);
    float*  hcat   = (float*)(ws + off);  off = align256(off + NHC * 4);
    bf16_t* zb     = (bf16_t*)(ws + off); off = align256(off + NH * 2);
    bf16_t* y1b    = (bf16_t*)(ws + off); off = align256(off + NH * 2);
    bf16_t* packed = (bf16_t*)(ws + off); off = align256(off + (size_t)6 * 16384 * 2);
    float*  stats  = (float*)(ws + off);  off = align256(off + 256 * 4);
    float*  pools  = (float*)(ws + off);  off = align256(off + (size_t)NUM_GRAPHS * 384 * 4);
    float*  counts = (float*)(ws + off);  off = align256(off + NUM_GRAPHS * 4);
    float*  gm1    = (float*)(ws + off);  off = align256(off + (size_t)NUM_GRAPHS * HIDDEN * 4);

    // Pack weights into WMMA B-operand layout (once per launch).
    pack_weights_kernel<<<(6 * 16384 + 255) / 256, 256, 0, stream>>>(W1, W2, packed);

    const int gemmBlocks = (N / 16 + 7) / 8;   // 782 blocks, 8 waves each

    for (int l = 0; l < LAYERS; ++l) {
        const float* hcur = (l == 0) ? x : h;

        // agg = 0
        zero_f32_kernel<<<((int)NH + 255) / 256, 256, 0, stream>>>(agg, (int)NH);
        // agg[dst] += h[src]
        scatter_add_kernel<<<(E * 32 + 255) / 256, 256, 0, stream>>>(hcur, ei, agg, E);
        // zb = bf16((1+eps)*h + agg)
        zin_kernel<<<((int)NH + 255) / 256, 256, 0, stream>>>(hcur, agg, eps_gin, l, zb, (int)NH);
        // y1b = relu(zb @ W1 + b1)       (bf16 out, feeds next GEMM)
        gemm128_wmma_kernel<true><<<gemmBlocks, 256, 0, stream>>>(
            zb, packed + (size_t)(l * 2 + 0) * 16384, b1 + l * HIDDEN, (void*)y1b, N);
        // z2 = relu(y1b @ W2 + b2)       (f32 out, feeds BN)
        gemm128_wmma_kernel<false><<<gemmBlocks, 256, 0, stream>>>(
            y1b, packed + (size_t)(l * 2 + 1) * 16384, b2 + l * HIDDEN, (void*)z2, N);
        // BN stats + apply
        zero_f32_kernel<<<1, 256, 0, stream>>>(stats, 256);
        bn_stats_kernel<<<512, HIDDEN, 0, stream>>>(z2, stats, N);
        bn_apply_kernel<<<((int)NH + 255) / 256, 256, 0, stream>>>(
            z2, stats, gamma, beta, l, h, hcat, N);
    }

    // Global mean pool over hcat
    const int poolN = NUM_GRAPHS * 384 + NUM_GRAPHS;
    zero_f32_kernel<<<(poolN + 255) / 256, 256, 0, stream>>>(pools, poolN); // pools+counts contiguous
    {
        long long total = (long long)N * 384;
        int blocks = (int)((total + 255) / 256);
        pool_acc_kernel<<<blocks, 256, 0, stream>>>(hcat, batch, pools, counts, N);
    }

    // Head MLP
    mlp1_kernel<<<NUM_GRAPHS, HIDDEN, 0, stream>>>(pools, counts, lin1_w, lin1_b, gm1);
    mlp2_kernel<<<NUM_GRAPHS, 32, 0, stream>>>(gm1, lin2_w, lin2_b, out);
}
